// LiquidGatedAttention_8847632629764
// MI455X (gfx1250) — matrix operand order repaired and checked by reading, not compiled, not run on hardware
//
#include <hip/hip_runtime.h>
#include <hip/hip_bf16.h>

// ---------------------------------------------------------------------------
// LiquidGatedAttention for MI455X (gfx1250, wave32, WMMA)
// B=2, T=2048, DIM=2048, H=16, HD=128, softcap=30
// ---------------------------------------------------------------------------

#define B_     2
#define T_     2048
#define DIM_   2048
#define BT_    (B_ * T_)        // 4096
#define NH_    16
#define HD_    128
#define SOFTCAP 30.0f
#define EPS_   1e-6f

typedef __attribute__((ext_vector_type(16))) __bf16       bf16x16;
typedef __attribute__((ext_vector_type(8)))  float        f32x8;
typedef __attribute__((ext_vector_type(4)))  unsigned int u32x4;
typedef int i32x4v __attribute__((vector_size(16)));   // matches builtin proto

union FragU { bf16x16 v; u32x4 q[2]; };
union ChunkU { u32x4 q; __bf16 h[8]; };

// --- CDNA5 async global->LDS copy (probe for builtins, fall back to sync) ---
#define AS1 __attribute__((address_space(1)))
#define AS3 __attribute__((address_space(3)))

#if defined(__has_builtin)
#if __has_builtin(__builtin_amdgcn_global_load_async_to_lds_b128) && \
    __has_builtin(__builtin_amdgcn_s_wait_asynccnt)
#define USE_ASYNC_LDS 1
#endif
#endif
#ifndef USE_ASYNC_LDS
#define USE_ASYNC_LDS 0
#endif

#if USE_ASYNC_LDS
// Builtin signature (verified via clang diagnostic): param0 = global int4*,
// param1 = LDS int4*, then imm offset + imm cpol. Generic LDS pointer ->
// 32-bit LDS address via integer detour (ISA 10.2: low 32 bits = LDS offset).
__device__ __forceinline__ AS3 i32x4v* lds_cast(void* p) {
  return (AS3 i32x4v*)(unsigned int)(unsigned long long)p;
}
__device__ __forceinline__ AS1 i32x4v* glb_cast(const void* p) {
  return (AS1 i32x4v*)(unsigned long long)p;
}
__device__ __forceinline__ void async_copy16(const void* g, void* l) {
  __builtin_amdgcn_global_load_async_to_lds_b128(glb_cast(g), lds_cast(l), 0, 0);
}
#endif

// Assemble a 32-byte WMMA operand from two 16-byte LDS chunks.
__device__ __forceinline__ bf16x16 load_frag(const __bf16* p0, const __bf16* p1) {
  FragU u;
  u.q[0] = *reinterpret_cast<const u32x4*>(p0);
  u.q[1] = *reinterpret_cast<const u32x4*>(p1);
  return u.v;
}

// ---------------------------------------------------------------------------
// fp32 -> bf16 conversion (weights and raw activations)
// ---------------------------------------------------------------------------
__global__ __launch_bounds__(256) void f32_to_bf16_kernel(
    const float* __restrict__ in, __bf16* __restrict__ out, int n) {
  int i = blockIdx.x * 256 + threadIdx.x;
  if (i < n) out[i] = (__bf16)in[i];
}

// ---------------------------------------------------------------------------
// RMSNorm: one block per row of x [BT, DIM]; output bf16 xn = x*rsqrt(ms)*w
// ---------------------------------------------------------------------------
__global__ __launch_bounds__(256) void rmsnorm_kernel(
    const float* __restrict__ x, const float* __restrict__ w,
    __bf16* __restrict__ xn) {
  __shared__ float red[256];
  const int tid = threadIdx.x;
  const size_t row = blockIdx.x;
  const float* xr = x + row * DIM_;
  float s = 0.f;
  #pragma unroll
  for (int i = tid; i < DIM_; i += 256) { float v = xr[i]; s += v * v; }
  red[tid] = s;
  __syncthreads();
  for (int st = 128; st > 0; st >>= 1) {
    if (tid < st) red[tid] += red[tid + st];
    __syncthreads();
  }
  const float inv = rsqrtf(red[0] * (1.0f / DIM_) + EPS_);
  #pragma unroll
  for (int i = tid; i < DIM_; i += 256)
    xn[row * DIM_ + i] = (__bf16)(xr[i] * inv * w[i]);
}

// ---------------------------------------------------------------------------
// NT GEMM: C[M,N] = A[M,K] * W[N,K]^T, bf16 in (K contiguous both operands).
// 128x128 block tile, K-slab 64, DOUBLE-BUFFERED LDS: next slab is fetched
// (async-to-LDS on CDNA5, register-batched otherwise) while WMMAs run on the
// current slab. 8 waves, each a 2x4 grid of v_wmma_f32_16x16x32_bf16.
// ---------------------------------------------------------------------------
#define GTK 64
#define GPAD 8   // LDS row padding (keeps 16B alignment, dodges bank conflicts)

__global__ __launch_bounds__(256, 2) void gemm_nt_bf16_kernel(
    const __bf16* __restrict__ A, const __bf16* __restrict__ W,
    void* __restrict__ Cout, int M, int N, int K, int store_bf16) {
  __shared__ __bf16 As[2][128][GTK + GPAD];
  __shared__ __bf16 Ws[2][128][GTK + GPAD];

  const int tid  = threadIdx.x;
  const int lane = tid & 31;
  const int wave = tid >> 5;
  const int wm   = wave >> 1;   // 0..3 : 32-row strip
  const int wn   = wave & 1;    // 0..1 : 64-col strip
  const int lg   = lane >> 4;   // lane half
  const int ll   = lane & 15;
  const int bm   = blockIdx.y * 128;
  const int bn   = blockIdx.x * 128;

  // Per-thread chunk coords: 4 chunks per matrix, rows crow+32*i, 16B col ccol
  const int crow = tid >> 3;
  const int ccol = (tid & 7) * 8;
  const __bf16* gA = A + (size_t)bm * K + ccol;
  const __bf16* gW = W + (size_t)bn * K + ccol;

  f32x8 acc[2][4];
  #pragma unroll
  for (int i = 0; i < 2; ++i)
    #pragma unroll
    for (int j = 0; j < 4; ++j) acc[i][j] = f32x8{};

#if USE_ASYNC_LDS
  auto stage_async = [&](int buf, int k0) {
    #pragma unroll
    for (int i = 0; i < 4; ++i) {
      const int row = crow + i * 32;
      async_copy16(gA + (size_t)row * K + k0, &As[buf][row][ccol]);
      async_copy16(gW + (size_t)row * K + k0, &Ws[buf][row][ccol]);
    }
  };
  stage_async(0, 0);
  __builtin_amdgcn_s_wait_asynccnt(0);
#else
  {
    u32x4 ra[4], rw[4];
    #pragma unroll
    for (int i = 0; i < 4; ++i) {   // batch all global loads (one wait total)
      ra[i] = *reinterpret_cast<const u32x4*>(gA + (size_t)(crow + i * 32) * K);
      rw[i] = *reinterpret_cast<const u32x4*>(gW + (size_t)(crow + i * 32) * K);
    }
    #pragma unroll
    for (int i = 0; i < 4; ++i) {
      *reinterpret_cast<u32x4*>(&As[0][crow + i * 32][ccol]) = ra[i];
      *reinterpret_cast<u32x4*>(&Ws[0][crow + i * 32][ccol]) = rw[i];
    }
  }
#endif
  __syncthreads();

  int buf = 0;
  for (int k0 = 0; k0 < K; k0 += GTK) {
    const bool more = (k0 + GTK) < K;
#if USE_ASYNC_LDS
    if (more) stage_async(buf ^ 1, k0 + GTK);   // overlaps with WMMAs below
#else
    u32x4 ra[4], rw[4];
    if (more) {                                  // issue loads before compute
      #pragma unroll
      for (int i = 0; i < 4; ++i) {
        ra[i] = *reinterpret_cast<const u32x4*>(gA + (size_t)(crow + i * 32) * K + k0 + GTK);
        rw[i] = *reinterpret_cast<const u32x4*>(gW + (size_t)(crow + i * 32) * K + k0 + GTK);
      }
      __builtin_prefetch(gA + (size_t)crow * K + k0 + 2 * GTK, 0, 3);
      __builtin_prefetch(gW + (size_t)crow * K + k0 + 2 * GTK, 0, 3);
    }
#endif

    #pragma unroll
    for (int ks = 0; ks < GTK; ks += 32) {
      bf16x16 af[2], bf[4];
      #pragma unroll
      for (int mi = 0; mi < 2; ++mi) {
        // A operand, this code's contiguous k order (matched in B): lane half lg holds 16 contiguous K {ks+lg*16..}
        const __bf16* p = &As[buf][wm * 32 + mi * 16 + ll][ks + lg * 16];
        af[mi] = load_frag(p, p + 8);
      }
      #pragma unroll
      for (int ni = 0; ni < 4; ++ni) {
        // B operand, same contiguous k order as A: column n = ll, lane half lg holds 16 contiguous K
        const __bf16* p = &Ws[buf][wn * 64 + ni * 16 + ll][ks + lg * 16];
        bf[ni] = load_frag(p, p + 8);
      }
      #pragma unroll
      for (int mi = 0; mi < 2; ++mi)
        #pragma unroll
        for (int ni = 0; ni < 4; ++ni)
          acc[mi][ni] = __builtin_amdgcn_wmma_f32_16x16x32_bf16(
              false, af[mi], false, bf[ni], (short)0, acc[mi][ni], false, false);
    }

#if USE_ASYNC_LDS
    if (more) __builtin_amdgcn_s_wait_asynccnt(0);
#else
    if (more) {
      #pragma unroll
      for (int i = 0; i < 4; ++i) {
        *reinterpret_cast<u32x4*>(&As[buf ^ 1][crow + i * 32][ccol]) = ra[i];
        *reinterpret_cast<u32x4*>(&Ws[buf ^ 1][crow + i * 32][ccol]) = rw[i];
      }
    }
#endif
    __syncthreads();
    buf ^= 1;
  }

  // Epilogue: C 16x16 f32 layout: VGPR r, lane half lg -> m = r + lg*8
  #pragma unroll
  for (int mi = 0; mi < 2; ++mi)
    #pragma unroll
    for (int ni = 0; ni < 4; ++ni) {
      const int mbase = bm + wm * 32 + mi * 16 + lg * 8;
      const int n     = bn + wn * 64 + ni * 16 + ll;
      #pragma unroll
      for (int r = 0; r < 8; ++r) {
        const float v   = acc[mi][ni][r];
        const size_t of = (size_t)(mbase + r) * N + n;
        if (store_bf16) ((__bf16*)Cout)[of] = (__bf16)v;
        else            ((float*)Cout)[of]  = v;
      }
    }
}

// ---------------------------------------------------------------------------
// Flash attention with tanh soft-cap + causal mask.
// Block = (q-tile of 128 rows, head, batch); 8 waves, wave owns 16 q rows.
// K blocks stream via async-to-LDS into double-buffered tiles; V is staged
// through registers (needs transpose) with its scatter deferred past compute.
// ---------------------------------------------------------------------------
#define QT 128
#define KT 64

__global__ __launch_bounds__(256, 1) void attention_kernel(
    const __bf16* __restrict__ Q, const __bf16* __restrict__ Km,
    const __bf16* __restrict__ V, __bf16* __restrict__ O) {
  __shared__ __bf16 Qs[QT][HD_ + 8];
  __shared__ __bf16 Ks[2][KT][HD_ + 8];
  __shared__ __bf16 Vt[2][HD_][KT + 8];   // transposed: [hd][key]
  __shared__ __bf16 Ps[8][16][KT + 8];    // per-wave probability tile

  const int tid  = threadIdx.x;
  const int lane = tid & 31;
  const int wave = tid >> 5;
  const int lg   = lane >> 4;
  const int ll   = lane & 15;
  const int qb   = blockIdx.x * QT;
  const int h    = blockIdx.y;
  const int b    = blockIdx.z;

  const size_t bo = (size_t)b * T_ * DIM_ + (size_t)h * HD_;
  const __bf16* Qg = Q + bo;
  const __bf16* Kg = Km + bo;
  const __bf16* Vg = V + bo;
  __bf16*       Og = O + bo;

  const int crow = tid >> 4;          // chunk row base (stride 16/64 rows)
  const int ccol = (tid & 15) * 8;    // chunk column (16B granules)

#if USE_ASYNC_LDS
  auto stage_K_async = [&](int buf, int k0) {
    #pragma unroll
    for (int i = 0; i < 4; ++i) {
      const int row = crow + i * 16;
      async_copy16(Kg + (size_t)(k0 + row) * DIM_ + ccol, &Ks[buf][row][ccol]);
    }
  };
#else
  auto stage_K_sync = [&](int buf, int k0) {
    u32x4 rk[4];
    #pragma unroll
    for (int i = 0; i < 4; ++i)
      rk[i] = *reinterpret_cast<const u32x4*>(Kg + (size_t)(k0 + crow + i * 16) * DIM_ + ccol);
    #pragma unroll
    for (int i = 0; i < 4; ++i)
      *reinterpret_cast<u32x4*>(&Ks[buf][crow + i * 16][ccol]) = rk[i];
  };
#endif
  auto load_V_regs = [&](u32x4 (&rv)[4], int k0) {
    #pragma unroll
    for (int i = 0; i < 4; ++i)
      rv[i] = *reinterpret_cast<const u32x4*>(Vg + (size_t)(k0 + crow + i * 16) * DIM_ + ccol);
  };
  auto scatter_V = [&](u32x4 (&rv)[4], int buf) {
    #pragma unroll
    for (int i = 0; i < 4; ++i) {
      ChunkU cu; cu.q = rv[i];
      const int row = crow + i * 16;
      #pragma unroll
      for (int e = 0; e < 8; ++e) Vt[buf][ccol + e][row] = cu.h[e];
    }
  };

  // ---- Prologue: stage Q tile + K/V block 0 ----
  u32x4 rv[4];
  load_V_regs(rv, 0);
#if USE_ASYNC_LDS
  #pragma unroll
  for (int i = 0; i < 8; ++i) {   // Q: 128x128 bf16, 8 chunks/thread
    const int row = crow + i * 16;
    async_copy16(Qg + (size_t)(qb + row) * DIM_ + ccol, &Qs[row][ccol]);
  }
  stage_K_async(0, 0);
  __builtin_amdgcn_s_wait_asynccnt(0);
#else
  #pragma unroll
  for (int i = 0; i < 8; ++i) {
    const int row = crow + i * 16;
    *reinterpret_cast<u32x4*>(&Qs[row][ccol]) =
        *reinterpret_cast<const u32x4*>(Qg + (size_t)(qb + row) * DIM_ + ccol);
  }
  stage_K_sync(0, 0);
#endif
  scatter_V(rv, 0);
  __syncthreads();

  // Preload this wave's Q fragments (4 K-steps over head dim).
  bf16x16 aQ[4];
  #pragma unroll
  for (int ks = 0; ks < 4; ++ks) {
    const __bf16* p = &Qs[wave * 16 + ll][ks * 32 + lg * 16];
    aQ[ks] = load_frag(p, p + 8);
  }

  f32x8 oacc[8];
  #pragma unroll
  for (int i = 0; i < 8; ++i) oacc[i] = f32x8{};
  float m_i[8], l_i[8];
  #pragma unroll
  for (int r = 0; r < 8; ++r) { m_i[r] = -1e30f; l_i[r] = 0.f; }

  const float scale = 0.08838834764831845f;  // 1/sqrt(128)
  const int kend = qb + QT;                  // causal: no keys past q tile

  int buf = 0;
  for (int k0 = 0; k0 < kend; k0 += KT) {
    const bool more = (k0 + KT) < kend;
    if (more) {                       // prefetch next K/V block during compute
#if USE_ASYNC_LDS
      stage_K_async(buf ^ 1, k0 + KT);
#endif
      load_V_regs(rv, k0 + KT);
    }

    // S = Q*K^T for this wave's 16 rows x 64 keys.
    f32x8 sacc[4];
    #pragma unroll
    for (int nt = 0; nt < 4; ++nt) sacc[nt] = f32x8{};
    #pragma unroll
    for (int nt = 0; nt < 4; ++nt)
      #pragma unroll
      for (int ks = 0; ks < 4; ++ks) {
        const __bf16* p = &Ks[buf][nt * 16 + ll][ks * 32 + lg * 16];
        bf16x16 bk = load_frag(p, p + 8);
        sacc[nt] = __builtin_amdgcn_wmma_f32_16x16x32_bf16(
            false, aQ[ks], false, bk, (short)0, sacc[nt], false, false);
      }

    // Soft-cap + causal mask + row max (16-lane reduction).
    float sv[4][8], rmax[8];
    #pragma unroll
    for (int r = 0; r < 8; ++r) rmax[r] = -1e30f;
    #pragma unroll
    for (int nt = 0; nt < 4; ++nt) {
      const int kg = k0 + nt * 16 + ll;
      #pragma unroll
      for (int r = 0; r < 8; ++r) {
        const int qg = qb + wave * 16 + lg * 8 + r;
        float s = sacc[nt][r] * scale;
        s = SOFTCAP * tanhf(s * (1.0f / SOFTCAP));
        if (kg > qg) s = -1e9f;
        sv[nt][r] = s;
        rmax[r] = fmaxf(rmax[r], s);
      }
    }
    #pragma unroll
    for (int r = 0; r < 8; ++r)
      #pragma unroll
      for (int m = 1; m < 16; m <<= 1)
        rmax[r] = fmaxf(rmax[r], __shfl_xor(rmax[r], m, 32));

    // Online softmax rescale.
    #pragma unroll
    for (int r = 0; r < 8; ++r) {
      const float mnew = fmaxf(m_i[r], rmax[r]);
      const float corr = expf(m_i[r] - mnew);
      m_i[r] = mnew;
      l_i[r] *= corr;
      #pragma unroll
      for (int d = 0; d < 8; ++d) oacc[d][r] *= corr;
    }

    // P = exp(s - m), stash bf16 tile, accumulate row sums.
    float psum[8];
    #pragma unroll
    for (int r = 0; r < 8; ++r) psum[r] = 0.f;
    #pragma unroll
    for (int nt = 0; nt < 4; ++nt)
      #pragma unroll
      for (int r = 0; r < 8; ++r) {
        const float p = expf(sv[nt][r] - m_i[r]);
        Ps[wave][lg * 8 + r][nt * 16 + ll] = (__bf16)p;
        psum[r] += p;
      }
    #pragma unroll
    for (int r = 0; r < 8; ++r) {
      #pragma unroll
      for (int m = 1; m < 16; m <<= 1) psum[r] += __shfl_xor(psum[r], m, 32);
      l_i[r] += psum[r];
    }

    // O += P * V  (K dim = 64 keys = 2 WMMA K-steps).
    #pragma unroll
    for (int ks = 0; ks < 2; ++ks) {
      const __bf16* pa = &Ps[wave][ll][ks * 32 + lg * 16];
      bf16x16 aP = load_frag(pa, pa + 8);
      #pragma unroll
      for (int nt = 0; nt < 8; ++nt) {
        const __bf16* pv = &Vt[buf][nt * 16 + ll][ks * 32 + lg * 16];
        bf16x16 bV = load_frag(pv, pv + 8);
        oacc[nt] = __builtin_amdgcn_wmma_f32_16x16x32_bf16(
            false, aP, false, bV, (short)0, oacc[nt], false, false);
      }
    }

    if (more) {
#if !USE_ASYNC_LDS
      stage_K_sync(buf ^ 1, k0 + KT);
#endif
      scatter_V(rv, buf ^ 1);
#if USE_ASYNC_LDS
      __builtin_amdgcn_s_wait_asynccnt(0);   // own asyncs done before barrier
#endif
    }
    __syncthreads();
    buf ^= 1;
  }

  // Normalize and store (bf16, [b*T+q][h*128+d] layout).
  #pragma unroll
  for (int nt = 0; nt < 8; ++nt)
    #pragma unroll
    for (int r = 0; r < 8; ++r) {
      const int qg = qb + wave * 16 + lg * 8 + r;
      Og[(size_t)qg * DIM_ + nt * 16 + ll] = (__bf16)(oacc[nt][r] / l_i[r]);
    }
}

// ---------------------------------------------------------------------------
// Fused gate epilogue: out = (1-sigmoid(gl+b))*x + sigmoid(gl+b)*proj
// ---------------------------------------------------------------------------
__global__ __launch_bounds__(256) void gate_epilogue_kernel(
    const float* __restrict__ x, const float* __restrict__ proj,
    const float* __restrict__ glog, const float* __restrict__ gbias,
    float* __restrict__ out, int n) {
  int i = blockIdx.x * 256 + threadIdx.x;
  if (i < n) {
    const float g = glog[i] + gbias[i & (DIM_ - 1)];
    const float a = 1.0f / (1.0f + expf(-g));
    out[i] = (1.0f - a) * x[i] + a * proj[i];
  }
}

// ---------------------------------------------------------------------------
// Host-side launch
// ---------------------------------------------------------------------------
extern "C" void kernel_launch(void* const* d_in, const int* in_sizes, int n_in,
                              void* d_out, int out_size, void* d_ws, size_t ws_size,
                              hipStream_t stream) {
  const float* x      = (const float*)d_in[0];
  // d_in[1] = additive mask (handled analytically as causal)
  const float* Wq     = (const float*)d_in[2];
  const float* Wk     = (const float*)d_in[3];
  const float* Wv     = (const float*)d_in[4];
  const float* Wo     = (const float*)d_in[5];
  const float* ln_w   = (const float*)d_in[6];
  const float* gate_W = (const float*)d_in[7];
  const float* gate_b = (const float*)d_in[8];
  float* out = (float*)d_out;

  // Workspace carve-up
  char* ws = (char*)d_ws;
  size_t off = 0;
  auto carve = [&](size_t bytes) { void* p = ws + off; off += bytes; return p; };
  const size_t ACT_BF16 = (size_t)BT_ * DIM_ * sizeof(__bf16);   // 16 MB
  const size_t W_BF16   = (size_t)DIM_ * DIM_ * sizeof(__bf16);  //  8 MB
  const size_t ACT_F32  = (size_t)BT_ * DIM_ * sizeof(float);    // 32 MB

  __bf16* xb   = (__bf16*)carve(ACT_BF16);
  __bf16* xn   = (__bf16*)carve(ACT_BF16);
  __bf16* wqb  = (__bf16*)carve(W_BF16);
  __bf16* wkb  = (__bf16*)carve(W_BF16);
  __bf16* wvb  = (__bf16*)carve(W_BF16);
  __bf16* wob  = (__bf16*)carve(W_BF16);
  __bf16* wgb  = (__bf16*)carve(W_BF16);
  __bf16* Qb   = (__bf16*)carve(ACT_BF16);
  __bf16* Kb   = (__bf16*)carve(ACT_BF16);
  __bf16* Vb   = (__bf16*)carve(ACT_BF16);
  __bf16* attn = (__bf16*)carve(ACT_BF16);
  float*  proj = (float*)carve(ACT_F32);
  float*  gate = (float*)carve(ACT_F32);
  (void)ws_size; (void)in_sizes; (void)n_in; (void)out_size;

  const int nW = DIM_ * DIM_;
  const int nA = BT_ * DIM_;

  // 1) bf16 conversions
  f32_to_bf16_kernel<<<(nA + 255) / 256, 256, 0, stream>>>(x, xb, nA);
  f32_to_bf16_kernel<<<(nW + 255) / 256, 256, 0, stream>>>(Wq, wqb, nW);
  f32_to_bf16_kernel<<<(nW + 255) / 256, 256, 0, stream>>>(Wk, wkb, nW);
  f32_to_bf16_kernel<<<(nW + 255) / 256, 256, 0, stream>>>(Wv, wvb, nW);
  f32_to_bf16_kernel<<<(nW + 255) / 256, 256, 0, stream>>>(Wo, wob, nW);
  f32_to_bf16_kernel<<<(nW + 255) / 256, 256, 0, stream>>>(gate_W, wgb, nW);

  // 2) RMSNorm
  rmsnorm_kernel<<<BT_, 256, 0, stream>>>(x, ln_w, xn);

  // 3) Q/K/V projections (WMMA NT GEMM, bf16 out)
  dim3 ggrid(DIM_ / 128, BT_ / 128);
  gemm_nt_bf16_kernel<<<ggrid, 256, 0, stream>>>(xn, wqb, Qb, BT_, DIM_, DIM_, 1);
  gemm_nt_bf16_kernel<<<ggrid, 256, 0, stream>>>(xn, wkb, Kb, BT_, DIM_, DIM_, 1);
  gemm_nt_bf16_kernel<<<ggrid, 256, 0, stream>>>(xn, wvb, Vb, BT_, DIM_, DIM_, 1);

  // 4) Flash attention (soft-cap + causal)
  dim3 agrid(T_ / QT, NH_, B_);
  attention_kernel<<<agrid, 256, 0, stream>>>(Qb, Kb, Vb, attn);

  // 5) Output projection + gate logits (fp32 out)
  gemm_nt_bf16_kernel<<<ggrid, 256, 0, stream>>>(attn, wob, proj, BT_, DIM_, DIM_, 0);
  gemm_nt_bf16_kernel<<<ggrid, 256, 0, stream>>>(xb,   wgb, gate, BT_, DIM_, DIM_, 0);

  // 6) Gated residual
  gate_epilogue_kernel<<<(nA + 255) / 256, 256, 0, stream>>>(x, proj, gate, gate_b, out, nA);
}
